// GraphMixer_59442347377029
// MI455X (gfx1250) — matrix operand order repaired and checked by reading, not compiled, not run on hardware
//
#include <hip/hip_runtime.h>

// ---------------------------------------------------------------------------
// GraphMixer on gfx1250 (MI455X): f16 WMMA (f32 accum) for all three GEMMs,
// fused per-node mixer keeping the [*,400] GELU intermediate in LDS.
// Round 3: correct B-operand layout per ISA (same split K order as A; sparse-B
// table in CDNA5 ISA), fast-path unguarded scatter epilogue.
// ---------------------------------------------------------------------------

#define K_SLOTS   20
#define HID       100
#define ROW_HIDN  400
#define EFEAT     128
#define TDIM      100
#define FEAT_PAD  256   // logical K pad: 228 -> 256 (8 k-chunks of 32)
#define HID_PAD   128   // logical K pad: 100 -> 128 (4 k-chunks)
#define RH_PAD    416   // logical K pad: 400 -> 416 (13 k-chunks)
#define NPAD      112   // N pad: 100 -> 112 (7 n-tiles)
// LDS strides: odd multiples of 8 halves -> 16B aligned, no bank conflicts
#define FS_LD     264   // f16 stride for 64x256 edge-feature tile
#define A1_LD     136   // f16 stride for 80x128 tile
#define A2_LD     424   // f16 stride for 80x416 tile
#define XS_LD     105   // f32 stride (odd -> conflict-free row access)

typedef __attribute__((ext_vector_type(16))) _Float16 v16h;
typedef __attribute__((ext_vector_type(8)))  _Float16 v8h;
typedef __attribute__((ext_vector_type(8)))  float    v8f;

__device__ __forceinline__ float gelu_f(float x) {
  return 0.5f * x * (1.0f + erff(x * 0.70710678118654752f));
}

// A-operand fragment (16x32 f16, rows are M). Source row-major [rows][ldk].
// ISA A layout: lanes 0-15 hold K = k0+0..7 and k0+16..23,
// lanes 16-31 hold K = k0+8..15 and k0+24..31 -> two 16B loads at +0/+16.
__device__ __forceinline__ v16h load_frag_a(const _Float16* p, int row0, int k0, int ldk) {
  const int l  = threadIdx.x & 31;
  const int r  = row0 + (l & 15);
  const int kk = k0 + ((l & 16) ? 8 : 0);
  const _Float16* q = p + r * ldk + kk;
  v8h lo = *(const v8h*)(q);
  v8h hi = *(const v8h*)(q + 16);
  v16h f;
#pragma unroll
  for (int i = 0; i < 8; ++i) { f[i] = lo[i]; f[i + 8] = hi[i]; }
  return f;
}

// B-operand fragment (32x16 f16, source holds B^T row-major [N][ldk]).
// B layout matches A: lane l -> column N = l&15;
// lanes 0-15 get K = k0+0..7 and k0+16..23,
// lanes 16-31 get K = k0+8..15 and k0+24..31 -> two 16B loads at +0/+16.
__device__ __forceinline__ v16h load_frag_b(const _Float16* p, int row0, int k0, int ldk) {
  const int l  = threadIdx.x & 31;
  const int r  = row0 + (l & 15);
  const int kk = k0 + ((l & 16) ? 8 : 0);
  const _Float16* q = p + r * ldk + kk;
  v8h lo = *(const v8h*)(q);
  v8h hi = *(const v8h*)(q + 16);
  v16h f;
#pragma unroll
  for (int i = 0; i < 8; ++i) { f[i] = lo[i]; f[i + 8] = hi[i]; }
  return f;
}

// ---------------------------------------------------------------------------
__global__ __launch_bounds__(256) void zero_kernel(float4* p, long long n4) {
  long long i = (long long)blockIdx.x * blockDim.x + threadIdx.x;
  long long stride = (long long)gridDim.x * blockDim.x;
  float4 z = make_float4(0.f, 0.f, 0.f, 0.f);
  for (; i < n4; i += stride) p[i] = z;
}

// Convert GEMM weights to f16 in B^T (N x K) layout, K zero-padded.
__global__ __launch_bounds__(256) void prep_kernel(
    const float* __restrict__ lin_w,   // [100][228]
    const float* __restrict__ row_w1,  // [400][100]
    const float* __restrict__ row_w2,  // [100][400]
    _Float16* __restrict__ linh,       // [112][256]
    _Float16* __restrict__ w1h,        // [400][128]
    _Float16* __restrict__ w2h)        // [112][416]
{
  int i = blockIdx.x * blockDim.x + threadIdx.x;
  int stride = gridDim.x * blockDim.x;
  for (int t = i; t < NPAD * FEAT_PAD; t += stride) {
    int r = t / FEAT_PAD, c = t % FEAT_PAD;
    float v = (r < HID && c < (EFEAT + TDIM)) ? lin_w[r * (EFEAT + TDIM) + c] : 0.f;
    linh[t] = (_Float16)v;
  }
  for (int t = i; t < ROW_HIDN * HID_PAD; t += stride) {
    int r = t / HID_PAD, c = t % HID_PAD;
    float v = (c < HID) ? row_w1[r * HID + c] : 0.f;
    w1h[t] = (_Float16)v;
  }
  for (int t = i; t < NPAD * RH_PAD; t += stride) {
    int r = t / RH_PAD, c = t % RH_PAD;
    float v = (r < HID && c < ROW_HIDN) ? row_w2[r * ROW_HIDN + c] : 0.f;
    w2h[t] = (_Float16)v;
  }
}

// ---------------------------------------------------------------------------
// Edge linear + time-encode + scatter. 64 edges per WG, 8 waves.
__global__ __launch_bounds__(256) void edge_kernel(
    const float* __restrict__ ef, const float* __restrict__ et,
    const int* __restrict__ nid, const int* __restrict__ sidx,
    const _Float16* __restrict__ linh, const float* __restrict__ lin_b,
    float* __restrict__ xbuf, int nE)
{
  __shared__ __align__(16) _Float16 fs[64 * FS_LD];
  __shared__ float ets[64];
  __shared__ int   pos[64];

  const int tid = threadIdx.x;
  const int e0  = blockIdx.x * 64;

  if (tid < 64) {
    int e = e0 + tid;
    ets[tid] = (e < nE) ? et[e] : 0.f;
    pos[tid] = (e < nE) ? (nid[e] * K_SLOTS + sidx[e]) : 0;
  }
  // edge features -> f16
  for (int t = tid; t < 64 * EFEAT; t += 256) {
    int e = t >> 7, c = t & 127;
    float v = ((e0 + e) < nE) ? ef[(long long)(e0 + e) * EFEAT + c] : 0.f;
    fs[e * FS_LD + c] = (_Float16)v;
  }
  __syncthreads();
  // time encode (cols 128..227) + zero pad (228..255)
  for (int t = tid; t < 64 * 128; t += 256) {
    int e = t >> 7, c = t & 127;
    float v = 0.f;
    if (c < TDIM) {
      float tw = __powf(10.f, -(9.f * (float)c) / 99.f);
      v = __cosf(ets[e] * tw);
    }
    fs[e * FS_LD + EFEAT + c] = (_Float16)v;
  }
  __syncthreads();

  const int wave = tid >> 5, l = tid & 31;
  const int nvalid = (nE - e0 < 64) ? (nE - e0) : 64;
  // 14 tasks = 7 n-tiles x 2 halves of the 4 M-tiles; B loaded once per task.
  for (int t = wave; t < 14; t += 8) {
    int nt = t >> 1, mh = t & 1;
    v16h b[8];
#pragma unroll
    for (int kc = 0; kc < 8; ++kc) b[kc] = load_frag_b(linh, nt * 16, kc * 32, FEAT_PAD);
    int n = nt * 16 + (l & 15);
    bool ok = n < HID;
    float bn = ok ? lin_b[n] : 0.f;
    for (int mt = mh * 2; mt < mh * 2 + 2; ++mt) {
      v8f acc = {};
#pragma unroll
      for (int kc = 0; kc < 8; ++kc) {
        v16h a = load_frag_a(fs, mt * 16, kc * 32, FS_LD);
        acc = __builtin_amdgcn_wmma_f32_16x16x32_f16(false, a, false, b[kc], (short)0, acc, false, false);
      }
      int mbase = mt * 16 + ((l & 16) ? 8 : 0);
      if (nvalid == 64) {           // uniform hot path: no per-row exec dance
        if (ok) {
#pragma unroll
          for (int r = 0; r < 8; ++r)
            xbuf[(long long)pos[mbase + r] * HID + n] = acc[r] + bn;
        }
      } else if (ok) {
#pragma unroll
        for (int r = 0; r < 8; ++r) {
          int m = mbase + r;
          if (m < nvalid)
            xbuf[(long long)pos[m] * HID + n] = acc[r] + bn;
        }
      }
    }
  }
}

// ---------------------------------------------------------------------------
// Fused mixer: 4 nodes per WG (80 rows = exactly 5 WMMA M-tiles), 8 waves.
__global__ __launch_bounds__(256) void mixer_kernel(
    const float* __restrict__ xbuf,
    const float* __restrict__ col_ln_g, const float* __restrict__ col_ln_b,
    const float* __restrict__ col_w1,   const float* __restrict__ col_b1,
    const float* __restrict__ col_w2,   const float* __restrict__ col_b2,
    const float* __restrict__ row_ln_g, const float* __restrict__ row_ln_b,
    const _Float16* __restrict__ w1h,   const float* __restrict__ row_b1,
    const _Float16* __restrict__ w2h,   const float* __restrict__ row_b2,
    const float* __restrict__ norm_g,   const float* __restrict__ norm_b,
    const float* __restrict__ out_w,    const float* __restrict__ out_b,
    float* __restrict__ out)
{
  extern __shared__ __align__(16) char smem[];
  float*    xs  = (float*)smem;               // [80][105] f32 residual tile
  float*    avg = xs + 80 * XS_LD;            // [4][105]
  float*    cw  = avg + 4 * XS_LD;            // 480 floats of col-mix params
  _Float16* a1  = (_Float16*)(cw + 480);      // [80][A1_LD] row-LN'd, f16
  _Float16* a2  = a1 + 80 * A1_LD;            // [80][A2_LD] gelu(H1), f16

  const int tid   = threadIdx.x;
  const int node0 = blockIdx.x * 4;
  const float* src = xbuf + (long long)node0 * K_SLOTS * HID;

  // ---- load x tile (80x100 contiguous f32) ----
  for (int t = tid; t < 80 * HID; t += 256) {
    int r = t / HID, c = t % HID;
    xs[r * XS_LD + c] = src[t];
  }
  // ---- stage col-mix params: [0,200) w1, [200,210) b1, [210,410) w2,
  //      [410,430) b2, [430,450) ln_g, [450,470) ln_b ----
  for (int t = tid; t < 470; t += 256) {
    float v;
    if      (t < 200) v = col_w1[t];
    else if (t < 210) v = col_b1[t - 200];
    else if (t < 410) v = col_w2[t - 210];
    else if (t < 430) v = col_b2[t - 410];
    else if (t < 450) v = col_ln_g[t - 430];
    else              v = col_ln_b[t - 450];
    cw[t] = v;
  }
  __syncthreads();

  // ---- column (token) mixing: 4 nodes x 100 channels, K=20 on VALU ----
  for (int t = tid; t < 400; t += 256) {
    int g = t / 100, c = t % 100;
    float v[K_SLOTS], ln[K_SLOTS], m1[10];
    float mu = 0.f;
#pragma unroll
    for (int k = 0; k < K_SLOTS; ++k) { v[k] = xs[(g * K_SLOTS + k) * XS_LD + c]; mu += v[k]; }
    mu *= 0.05f;
    float var = 0.f;
#pragma unroll
    for (int k = 0; k < K_SLOTS; ++k) { float d = v[k] - mu; var += d * d; }
    var *= 0.05f;
    float rs = rsqrtf(var + 1e-5f);
#pragma unroll
    for (int k = 0; k < K_SLOTS; ++k) ln[k] = (v[k] - mu) * rs * cw[430 + k] + cw[450 + k];
#pragma unroll
    for (int j = 0; j < 10; ++j) {
      float s = cw[200 + j];
#pragma unroll
      for (int k = 0; k < K_SLOTS; ++k) s += cw[j * K_SLOTS + k] * ln[k];
      m1[j] = gelu_f(s);
    }
#pragma unroll
    for (int k = 0; k < K_SLOTS; ++k) {
      float s = cw[410 + k];
#pragma unroll
      for (int j = 0; j < 10; ++j) s += cw[210 + k * 10 + j] * m1[j];
      xs[(g * K_SLOTS + k) * XS_LD + c] = v[k] + s;
    }
  }
  __syncthreads();

  // ---- row LayerNorm -> f16 A1, zero pads of A1 and A2 ----
  for (int r = tid; r < 80; r += 256) {
    float mu = 0.f;
    for (int c = 0; c < HID; ++c) mu += xs[r * XS_LD + c];
    mu *= 0.01f;
    float var = 0.f;
    for (int c = 0; c < HID; ++c) { float d = xs[r * XS_LD + c] - mu; var += d * d; }
    var *= 0.01f;
    float rs = rsqrtf(var + 1e-5f);
    for (int c = 0; c < HID; ++c)
      a1[r * A1_LD + c] = (_Float16)((xs[r * XS_LD + c] - mu) * rs * row_ln_g[c] + row_ln_b[c]);
    for (int c = HID; c < HID_PAD; ++c)      a1[r * A1_LD + c] = (_Float16)0.f;
    for (int c = ROW_HIDN; c < RH_PAD; ++c)  a2[r * A2_LD + c] = (_Float16)0.f;
  }
  __syncthreads();

  const int wave = tid >> 5, l = tid & 31;

  // ---- GEMM1: [80x128] @ [128x400] -> gelu -> A2.
  //      One N-tile per wave iteration; B register-resident across 5 M-tiles.
  for (int nt = wave; nt < 25; nt += 8) {
    v16h b[4];
#pragma unroll
    for (int kc = 0; kc < 4; ++kc) b[kc] = load_frag_b(w1h, nt * 16, kc * 32, HID_PAD);
    int n = nt * 16 + (l & 15);
    float b1v = row_b1[n];
#pragma unroll
    for (int mt = 0; mt < 5; ++mt) {
      v8f acc = {};
#pragma unroll
      for (int kc = 0; kc < 4; ++kc) {
        v16h a = load_frag_a(a1, mt * 16, kc * 32, A1_LD);
        acc = __builtin_amdgcn_wmma_f32_16x16x32_f16(false, a, false, b[kc], (short)0, acc, false, false);
      }
      int mbase = mt * 16 + ((l & 16) ? 8 : 0);
#pragma unroll
      for (int r = 0; r < 8; ++r)
        a2[(mbase + r) * A2_LD + n] = (_Float16)gelu_f(acc[r] + b1v);
    }
  }
  __syncthreads();

  // ---- GEMM2: [80x416] @ [416x100] + residual into xs.
  //      14 tasks = 7 n-tiles x 2 M-halves; 13 B-frags register-resident.
  for (int t = wave; t < 14; t += 8) {
    int nt = t >> 1, mh = t & 1;
    v16h b[13];
#pragma unroll
    for (int kc = 0; kc < 13; ++kc) b[kc] = load_frag_b(w2h, nt * 16, kc * 32, RH_PAD);
    int n = nt * 16 + (l & 15);
    bool ok = n < HID;
    float b2v = ok ? row_b2[n] : 0.f;
    int mlo = mh ? 3 : 0, mhi = mh ? 5 : 3;
    for (int mt = mlo; mt < mhi; ++mt) {
      v8f acc = {};
#pragma unroll
      for (int kc = 0; kc < 13; ++kc) {
        v16h a = load_frag_a(a2, mt * 16, kc * 32, A2_LD);
        acc = __builtin_amdgcn_wmma_f32_16x16x32_f16(false, a, false, b[kc], (short)0, acc, false, false);
      }
      if (ok) {
        int mbase = mt * 16 + ((l & 16) ? 8 : 0);
#pragma unroll
        for (int r = 0; r < 8; ++r)
          xs[(mbase + r) * XS_LD + n] += acc[r] + b2v;
      }
    }
  }
  __syncthreads();

  // ---- final LayerNorm per row (in place) ----
  for (int r = tid; r < 80; r += 256) {
    float mu = 0.f;
    for (int c = 0; c < HID; ++c) mu += xs[r * XS_LD + c];
    mu *= 0.01f;
    float var = 0.f;
    for (int c = 0; c < HID; ++c) { float d = xs[r * XS_LD + c] - mu; var += d * d; }
    var *= 0.01f;
    float rs = rsqrtf(var + 1e-5f);
    for (int c = 0; c < HID; ++c)
      xs[r * XS_LD + c] = (xs[r * XS_LD + c] - mu) * rs * norm_g[c] + norm_b[c];
  }
  __syncthreads();

  // ---- mean over K slots ----
  for (int t = tid; t < 400; t += 256) {
    int g = t / 100, c = t % 100;
    float s = 0.f;
#pragma unroll
    for (int k = 0; k < K_SLOTS; ++k) s += xs[(g * K_SLOTS + k) * XS_LD + c];
    avg[g * XS_LD + c] = s * 0.05f;
  }
  __syncthreads();

  // ---- output projection [100]->[100] per node ----
  for (int t = tid; t < 400; t += 256) {
    int g = t / 100, oc = t % 100;
    float s = out_b[oc];
    const float* w = out_w + oc * HID;
    for (int c = 0; c < HID; ++c) s += w[c] * avg[g * XS_LD + c];
    out[(long long)(node0 + g) * HID + oc] = s;
  }
}

// ---------------------------------------------------------------------------
extern "C" void kernel_launch(void* const* d_in, const int* in_sizes, int n_in,
                              void* d_out, int out_size, void* d_ws, size_t ws_size,
                              hipStream_t stream) {
  const float* ef       = (const float*)d_in[0];
  const float* et       = (const float*)d_in[1];
  const int*   nid      = (const int*)d_in[2];
  const int*   sidx     = (const int*)d_in[3];
  const float* lin_w    = (const float*)d_in[4];
  const float* lin_b    = (const float*)d_in[5];
  const float* col_ln_g = (const float*)d_in[6];
  const float* col_ln_b = (const float*)d_in[7];
  const float* col_w1   = (const float*)d_in[8];
  const float* col_b1   = (const float*)d_in[9];
  const float* col_w2   = (const float*)d_in[10];
  const float* col_b2   = (const float*)d_in[11];
  const float* row_ln_g = (const float*)d_in[12];
  const float* row_ln_b = (const float*)d_in[13];
  const float* row_w1   = (const float*)d_in[14];
  const float* row_b1   = (const float*)d_in[15];
  const float* row_w2   = (const float*)d_in[16];
  const float* row_b2   = (const float*)d_in[17];
  const float* norm_g   = (const float*)d_in[18];
  const float* norm_b   = (const float*)d_in[19];
  const float* out_w    = (const float*)d_in[20];
  const float* out_b    = (const float*)d_in[21];
  float* out = (float*)d_out;

  const int nE     = in_sizes[1];
  const int nNodes = out_size / HID;        // 20000

  char* ws = (char*)d_ws;
  float* xbuf = (float*)ws;                                     // [N*K][100] f32
  size_t off = (size_t)nNodes * K_SLOTS * HID * sizeof(float);
  _Float16* linh = (_Float16*)(ws + off); off += (size_t)NPAD * FEAT_PAD * 2;
  _Float16* w1h  = (_Float16*)(ws + off); off += (size_t)ROW_HIDN * HID_PAD * 2;
  _Float16* w2h  = (_Float16*)(ws + off); off += (size_t)NPAD * RH_PAD * 2;

  long long n4 = (long long)nNodes * K_SLOTS * HID / 4;
  zero_kernel<<<4096, 256, 0, stream>>>((float4*)xbuf, n4);
  prep_kernel<<<256, 256, 0, stream>>>(lin_w, row_w1, row_w2, linh, w1h, w2h);
  edge_kernel<<<(nE + 63) / 64, 256, 0, stream>>>(ef, et, nid, sidx, linh, lin_b, xbuf, nE);

  int shmem = 80 * XS_LD * 4 + 4 * XS_LD * 4 + 480 * 4
            + 80 * A1_LD * 2 + 80 * A2_LD * 2;                  // 126,800 B
  hipFuncSetAttribute((const void*)mixer_kernel,
                      hipFuncAttributeMaxDynamicSharedMemorySize, shmem);
  mixer_kernel<<<nNodes / 4, 256, shmem, stream>>>(
      xbuf, col_ln_g, col_ln_b, col_w1, col_b1, col_w2, col_b2,
      row_ln_g, row_ln_b, w1h, row_b1, w2h, row_b2,
      norm_g, norm_b, out_w, out_b, out);
}
